// Upsample_25013889532341
// MI455X (gfx1250) — matrix operand order repaired and checked by reading, not compiled, not run on hardware
//
#include <hip/hip_runtime.h>

// Fused zero-stuff-upsample(x2) + 17-tap sinc-Kaiser FIR + 1x1 channel mix
// for MI455X (gfx1250, wave32, WMMA).
//
// Algebra: out = FIR(W @ x) (filter commutes with the per-channel-identical
// depthwise conv). Polyphase: even taps of the 17-tap kernel are exactly 0
// except the center (sinc at integers), so:
//   out[..., 2s]   = C0 * y[..., s]
//   out[..., 2s+1] = sum_{c=0..7} KO[c] * y[..., s-3+c]
// with y = W @ x (reflection maps even up-domain indices to even indices, so
// x-domain reflection m<0 -> -m, m>=n -> 2n-1-m is exact).
//
// Precision: bf16 hi/lo split GEMM (A_hi*B_hi + A_hi*B_lo + A_lo*B_hi) with
// f32 accumulation -> ~2^-16 relative error at bf16-WMMA rates. The problem
// is HBM-bound (~256 MB mandatory traffic @ 23.3 TB/s ~= 11 us); x (128 MB)
// is L2-resident (192 MB L2), so inter-workgroup halo re-reads are L2 hits
// and output stores are issued non-temporal to preserve x residency.

typedef __attribute__((ext_vector_type(16))) __bf16 v16bf;
typedef __attribute__((ext_vector_type(8)))  float  v8f;
typedef __attribute__((ext_vector_type(2)))  float  f32x2;

#define NSAMP  16384   // x samples per channel
#define DCH    256     // input channels  (GEMM K)
#define OCH    128     // output channels (GEMM M)
#define SBLK   32      // y-samples produced per workgroup (-> 64 output t)
#define YH     48      // staged y-tile width incl. 8-sample halo each side
#define NT     3       // WMMA N-tiles (48 / 16)
#define KSTEPS 8       // 256 / 32

__device__ __forceinline__ __bf16 bf16_hi(float f, float* back) {
  // truncate to bf16 (hi part); remainder goes to the lo bf16 via RNE convert
  unsigned u = __builtin_bit_cast(unsigned, f) & 0xFFFF0000u;
  *back = __builtin_bit_cast(float, u);
  unsigned short s = (unsigned short)(u >> 16);
  return __builtin_bit_cast(__bf16, s);
}

__global__ __launch_bounds__(256) void upfir_gemm_wmma(
    const float* __restrict__ x, const float* __restrict__ Wm,
    float* __restrict__ out) {
  // 48 KB static LDS; Y (25 KB) aliases the B staging buffers after a barrier.
  __shared__ __align__(64) unsigned char smem[49152];
  __bf16* Bh = (__bf16*)smem;                 // [KSTEPS][NT][32 lanes][16] bf16 hi
  __bf16* Bl = Bh + KSTEPS * NT * 32 * 16;    // same, lo part
  float*  Y  = (float*)smem;                  // [128][49] f32, phase 3+

  const int tid = threadIdx.x;
  const int blk = blockIdx.x;
  const int b   = blk >> 9;                   // 512 s-blocks per batch
  const int sb  = blk & 511;
  const int s0  = sb * SBLK;
  const long t0 = (long)s0 * 2;

  // ---- Phase 1: stage x tile (reflect-padded) as bf16 hi/lo, already in the
  // B(32x16) fragment order used here (contiguous K, A matches): lanes 0-15 hold K=0..15 of column
  // `lane`, lanes 16-31 hold K=16..31 of column `lane-16`; element h == K%16.
  for (int i = tid; i < DCH * YH; i += 256) {
    int d = i / YH, c = i % YH;               // channel, tile column
    int gs = s0 - 8 + c;                      // sample index with halo
    gs = (gs < 0) ? -gs : gs;                 // reflect left
    gs = (gs >= NSAMP) ? (2 * NSAMP - 1 - gs) : gs;  // reflect right
    float v = x[((size_t)b * DCH + d) * NSAMP + gs];
    float hif;
    __bf16 h = bf16_hi(v, &hif);
    __bf16 l = (__bf16)(v - hif);
    int k    = d >> 5;                        // K-step
    int dr   = d & 31;                        // K within step
    int lane = (c & 15) | (dr & 16);
    int hh   = dr & 15;
    int nt   = c >> 4;
    int idx  = (((k * NT + nt) * 32 + lane) << 4) + hh;
    Bh[idx] = h;
    Bl[idx] = l;
  }
  __syncthreads();

  // ---- Phase 2: y = W @ x_tile via split-bf16 WMMA, f32 accumulation.
  const int w      = tid >> 5;                // wave id (wave32)
  const int lane   = tid & 31;
  const int row    = (w << 4) + (lane & 15);  // output channel (A-matrix M row)
  const int kshift = (lane & 16) ? 8 : 0;     // A layout: lanes 16-31 K-offset

  // Preload A fragments (hi/lo) for this wave's 16 output channels.
  // 16-bit A(16x32) layout (matches B staging): lane<16: K=h; lanes>=16: K=h+16.
  v16bf ah[KSTEPS], al[KSTEPS];
#pragma unroll
  for (int k = 0; k < KSTEPS; ++k) {
#pragma unroll
    for (int h = 0; h < 16; ++h) {
      int K = (k << 5) + 2 * kshift + h;
      float v = Wm[row * DCH + K];
      float hif;
      ah[k][h] = bf16_hi(v, &hif);
      al[k][h] = (__bf16)(v - hif);
    }
  }

  v8f acc[NT];
#pragma unroll
  for (int nt = 0; nt < NT; ++nt) {
    v8f a = {};
#pragma unroll
    for (int k = 0; k < KSTEPS; ++k) {
      const v16bf bh = *(const v16bf*)(Bh + (((k * NT + nt) * 32 + lane) << 4));
      const v16bf bl = *(const v16bf*)(Bl + (((k * NT + nt) * 32 + lane) << 4));
      a = __builtin_amdgcn_wmma_f32_16x16x32_bf16(false, ah[k], false, bh,
                                                  (short)0, a, false, false);
      a = __builtin_amdgcn_wmma_f32_16x16x32_bf16(false, ah[k], false, bl,
                                                  (short)0, a, false, false);
      a = __builtin_amdgcn_wmma_f32_16x16x32_bf16(false, al[k], false, bh,
                                                  (short)0, a, false, false);
    }
    acc[nt] = a;
  }
  __syncthreads();  // everyone is done reading B; Y may alias it now

  // ---- Phase 3: spill y tile to LDS (C/D layout: VGPR v -> M=v(+8), N=lane%16)
#pragma unroll
  for (int nt = 0; nt < NT; ++nt) {
#pragma unroll
    for (int v = 0; v < 8; ++v) {
      int r   = (w << 4) + v + ((lane & 16) ? 8 : 0);
      int col = nt * 16 + (lane & 15);
      Y[r * 49 + col] = acc[nt][v];
    }
  }
  __syncthreads();

  // ---- Phase 4: polyphase FIR. Each lane owns sample-pair sp == lane:
  // computes BOTH phases (even = C0*y[sp], odd = 8-tap) and stores them as one
  // float2 -> global_store_b64, 256 B contiguous per wave-store, no divergence.
  // Stores are non-temporal: out is write-once, keep L2 for x.
  const float C0 = 0.49989363f;  // center tap / kernel sum
  const float KO[8] = {-0.0026871f,  0.0184400f, -0.0699829f, 0.3042832f,
                        0.3042832f, -0.0699829f,  0.0184400f, -0.0026871f};
#pragma unroll 4
  for (int r = 0; r < 16; ++r) {
    int o = (w << 4) + r;                      // this wave's output channels
    const float* yr = Y + o * 49 + 8 + lane;   // col 8 == sample s0
    f32x2 v2;
    v2.x = C0 * yr[0];                         // even phase: t = 2*lane
    float odd = 0.f;                           // odd phase:  t = 2*lane + 1
#pragma unroll
    for (int c2 = 0; c2 < 8; ++c2) odd += KO[c2] * yr[c2 - 3];
    v2.y = odd;
    f32x2* dst = (f32x2*)(out + ((size_t)(b * OCH + o)) * (size_t)(2 * NSAMP) +
                          (size_t)t0 + 2 * lane);
    __builtin_nontemporal_store(v2, dst);
  }
}

extern "C" void kernel_launch(void* const* d_in, const int* in_sizes, int n_in,
                              void* d_out, int out_size, void* d_ws,
                              size_t ws_size, hipStream_t stream) {
  (void)in_sizes; (void)n_in; (void)out_size; (void)d_ws; (void)ws_size;
  const float* x  = (const float*)d_in[0];   // [8, 256, 16384] f32
  const float* Wm = (const float*)d_in[1];   // [128, 256] f32
  float* out = (float*)d_out;                // [8, 128, 32768] f32

  dim3 grid(8 * (NSAMP / SBLK));             // 4096 workgroups
  dim3 block(256);                           // 8 wave32 waves
  upfir_gemm_wmma<<<grid, block, 0, stream>>>(x, Wm, out);
}
